// OuterProductMean_23562190586478
// MI455X (gfx1250) — matrix operand order repaired and checked by reading, not compiled, not run on hardware
//
#include <hip/hip_runtime.h>

typedef _Float16 v16h __attribute__((ext_vector_type(16)));
typedef _Float16 v8h  __attribute__((ext_vector_type(8)));
typedef float    v8f  __attribute__((ext_vector_type(8)));
typedef int      v4ig __attribute__((vector_size(16)));   // GCC-style, matches builtin proto

#define S_DIM 512
#define N_DIM 256
#define CM    256
#define CH    32
#define CZ    128

#if __has_builtin(__builtin_amdgcn_global_load_async_to_lds_b128)
#define USE_ASYNC_LDS 1
#else
#define USE_ASYNC_LDS 0
#endif

// ---------- WMMA fragment helpers (contiguous k = 16*(lane/16) + s, same for A and B) ----------
static __device__ __forceinline__ v16h cat8(v8h lo, v8h hi) {
  v16h r;
#pragma unroll
  for (int e = 0; e < 8; ++e) { r[e] = lo[e]; r[8 + e] = hi[e]; }
  return r;
}
// A-matrix 16x32 f16 fragment. `row` points at this lane's M-row, k-chunk base.
// lanes 0-15: elems 0..15 = K 0..15 (kbase = 0)
// lanes 16-31: elems 0..15 = K 16..31 (kbase = 16)
static __device__ __forceinline__ v16h ldfragA(const _Float16* row, int kbase) {
  v8h lo = *(const v8h*)(row + kbase);
  v8h hi = *(const v8h*)(row + kbase + 8);
  return cat8(lo, hi);
}
// B-matrix 32x16 f16 fragment: lane holds column N=lane&15, K = kb..kb+15
// contiguous (kb = (lane>>4)*16). `p` points at those 16 contiguous halves.
static __device__ __forceinline__ v16h ldfragB(const _Float16* p) {
  v8h lo = *(const v8h*)p;
  v8h hi = *(const v8h*)(p + 8);
  return cat8(lo, hi);
}

static __device__ __forceinline__ void wait_async0() {
#if __has_builtin(__builtin_amdgcn_s_wait_asynccnt)
  __builtin_amdgcn_s_wait_asynccnt(0);
#else
  asm volatile("s_wait_asynccnt 0x0" ::: "memory");
#endif
}

#if USE_ASYNC_LDS
// Copy 32 bytes global -> LDS per lane via GLOBAL_LOAD_ASYNC_TO_LDS_B128
// (tracked on ASYNCcnt; INST_OFFSET applies to both src and dst).
static __device__ __forceinline__ void async_cp32(const _Float16* g, _Float16* l) {
  typedef __attribute__((address_space(1))) v4ig* gp_t;
  typedef __attribute__((address_space(3))) v4ig* lp_t;
  gp_t gs = (gp_t)g;
  lp_t ls = (lp_t)l;
  __builtin_amdgcn_global_load_async_to_lds_b128(gs, ls, 0, 0);
  __builtin_amdgcn_global_load_async_to_lds_b128(gs, ls, 32, 0);
}
#endif

// ---------- Kernel 1: norm[i][j] = sum_s mask[s,i]*mask[s,j]; a_norm[i] ----
__global__ void __launch_bounds__(256)
opm_norm_kernel(const float* __restrict__ mask,
                float* __restrict__ normW,
                float* __restrict__ a_normW) {
  __shared__ float colI[S_DIM];
  __shared__ float red[256];
  const int i = blockIdx.x, tid = threadIdx.x;
  colI[tid]       = mask[tid * N_DIM + i];
  colI[tid + 256] = mask[(tid + 256) * N_DIM + i];
  __syncthreads();
  float acc = 0.f;
#pragma unroll 4
  for (int s = 0; s < S_DIM; ++s) acc += colI[s] * mask[s * N_DIM + tid];
  normW[i * N_DIM + tid] = acc;
  red[tid] = acc;
  __syncthreads();
  for (int o = 128; o > 0; o >>= 1) {
    if (tid < o) red[tid] += red[tid + o];
    __syncthreads();
  }
  if (tid == 0) a_normW[i] = red[0] * (1.0f / (float)N_DIM);
}

// ---------- Kernel 2: w_out f32 -> f16 ------------------------------------
__global__ void __launch_bounds__(256)
opm_wout_cvt_kernel(const float* __restrict__ w_out, _Float16* __restrict__ woutH) {
  const int k = blockIdx.x * 256 + threadIdx.x;
  woutH[k] = (_Float16)w_out[k];
}

// ---------- Kernel 3: fused LN + dual projection + mask + scaling ---------
// Emits atT[n][h][s], btT[n][h][s] in f16 (K=s contiguous for GEMM1 frags).
#define W_LD 264
__global__ void __launch_bounds__(256)
opm_lnproj_kernel(const float* __restrict__ m,
                  const float* __restrict__ mask,
                  const float* __restrict__ ln_w,
                  const float* __restrict__ ln_b,
                  const float* __restrict__ w1,
                  const float* __restrict__ b1,
                  const float* __restrict__ w2,
                  const float* __restrict__ b2,
                  const float* __restrict__ a_normW,
                  _Float16* __restrict__ atT,
                  _Float16* __restrict__ btT) {
  __shared__ _Float16 w1h[CH * W_LD];
  __shared__ _Float16 w2h[CH * W_LD];
  __shared__ float mnbuf[8 * CM];
  const int tid = threadIdx.x;
  for (int k = tid; k < CH * CM; k += 256) {
    const int h = k >> 8, c = k & 255;
    w1h[h * W_LD + c] = (_Float16)w1[k];
    w2h[h * W_LD + c] = (_Float16)w2[k];
  }
  __syncthreads();
  const int wave = tid >> 5, lane = tid & 31;
  float* mnw = mnbuf + wave * CM;
  const int rowbase = blockIdx.x * 128 + wave * 16;
  for (int r = 0; r < 16; ++r) {
    const int t = rowbase + r;         // flat (s*N + n)
    const int s = t >> 8, n = t & 255;
    const float* mrow = m + (size_t)t * CM;
    const float4 x0 = ((const float4*)mrow)[lane * 2];
    const float4 x1 = ((const float4*)mrow)[lane * 2 + 1];
    float xs[8] = {x0.x, x0.y, x0.z, x0.w, x1.x, x1.y, x1.z, x1.w};
    float sum = 0.f, sq = 0.f;
#pragma unroll
    for (int e = 0; e < 8; ++e) { sum += xs[e]; sq += xs[e] * xs[e]; }
#pragma unroll
    for (int o = 16; o >= 1; o >>= 1) {
      sum += __shfl_xor(sum, o);
      sq  += __shfl_xor(sq, o);
    }
    const float mu = sum * (1.0f / CM);
    const float var = sq * (1.0f / CM) - mu * mu;
    const float rstd = rsqrtf(var + 1e-5f);
    const int c0 = lane * 8;
#pragma unroll
    for (int e = 0; e < 8; ++e)
      mnw[c0 + e] = (xs[e] - mu) * rstd * ln_w[c0 + e] + ln_b[c0 + e];
    __builtin_amdgcn_wave_barrier();   // wave-coherent LDS publish
    const int h = lane;
    float acc1 = b1[h], acc2 = b2[h];
#pragma unroll 4
    for (int c = 0; c < CM; ++c) {
      const float v = mnw[c];
      acc1 += v * (float)w1h[h * W_LD + c];
      acc2 += v * (float)w2h[h * W_LD + c];
    }
    __builtin_amdgcn_wave_barrier();   // before mnw reuse next iteration
    const float mv = mask[t];
    const float sa = mv / (a_normW[n] + 1e-4f);
    atT[((size_t)n * CH + h) * S_DIM + s] = (_Float16)(acc1 * sa);
    btT[((size_t)n * CH + h) * S_DIM + s] = (_Float16)(acc2 * mv);
  }
}

// ---------- Kernel 4: fused outer-product GEMM + w_out contraction --------
// Block: 4 i's x 4 j's. GEMM1: (128 x 128 x 512) f16 WMMA -> O' (f32 regs),
// double-buffered LDS with async global->LDS copies overlapped with WMMA.
// O' -> LDS f16 (aliasing the dead A/B buffers), GEMM2: (16 x 128 x 1024),
// scaled epilogue.
#define AS_LD 40                 // 32 K-halves + 8 pad
#define OS_LD 136                // 128 + 8 pad
#define BUF_HALFS (128 * AS_LD)  // 5120 halves = 10 KB per As or Bs buffer
#define NCHUNK (S_DIM / 32)      // 16
__global__ void __launch_bounds__(256)
opm_gemm_kernel(const _Float16* __restrict__ atT,
                const _Float16* __restrict__ btT,
                const _Float16* __restrict__ woutH,
                const float* __restrict__ normW,
                const float* __restrict__ a_normW,
                const float* __restrict__ b_out,
                float* __restrict__ out) {
  // 4 buffers (As0,Bs0,As1,Bs1) = 40 KB; O' staging (34 KB) aliases them.
  // Buffer `c` (c in {0,1}): As = smem + c*2*BUF_HALFS, Bs = As + BUF_HALFS.
  __shared__ __align__(16) _Float16 smem[4 * BUF_HALFS];
  __shared__ float sD[16], sNB[16];
  _Float16* Os = smem;  // 128*OS_LD = 17408 halves <= 20480 halves

  const int tid = threadIdx.x;
  const int wave = tid >> 5, lane = tid & 31;
  const int i0 = blockIdx.x * 4, j0 = blockIdx.y * 4;

  if (tid < 16) {
    const int i = i0 + (tid >> 2), j = j0 + (tid & 3);
    const float nrm = normW[i * N_DIM + j];
    const float an = a_normW[i];
    sD[tid]  = 1.0f / (nrm / (an + 1e-7f) + 1e-4f);
    sNB[tid] = 1.0f / (nrm + 0.001f);
  }

  // wave -> 2 m-tiles x 4 n-tiles of the 128x128 O' block
  const int wm = wave & 3, wn = wave >> 2;
  v8f acc[2][4] = {};

  // cooperative loader: 2 threads per row, 32 bytes each
  const int ldrow = tid >> 1;
  const int ldpart = (tid & 1) * 16;   // halves
  const int ldoff = ldrow * AS_LD + ldpart;
  const _Float16* gA = atT + ((size_t)(i0 * CH) + ldrow) * S_DIM + ldpart;
  const _Float16* gB = btT + ((size_t)(j0 * CH) + ldrow) * S_DIM + ldpart;

  const int mrow  = lane & 15;
  const int kbase = (lane >> 4) * 16;
  const int kbB   = (lane >> 4) * 16;

#if USE_ASYNC_LDS
  // ---- async pipeline: issue chunk kc+1 while computing chunk kc ----
  async_cp32(gA, smem + ldoff);
  async_cp32(gB, smem + BUF_HALFS + ldoff);
  for (int kc = 0; kc < NCHUNK; ++kc) {
    const int cur = kc & 1;
    wait_async0();        // own async copies landed in LDS
    __syncthreads();      // everyone's copies visible; prior readers done
    if (kc + 1 < NCHUNK) {
      _Float16* Asn = smem + (cur ^ 1) * 2 * BUF_HALFS;
      async_cp32(gA + (kc + 1) * 32, Asn + ldoff);
      async_cp32(gB + (kc + 1) * 32, Asn + BUF_HALFS + ldoff);
      __builtin_prefetch(gA + (kc + 2) * 32, 0, 3);
      __builtin_prefetch(gB + (kc + 2) * 32, 0, 3);
    }
    const _Float16* As = smem + cur * 2 * BUF_HALFS;
    const _Float16* Bs = As + BUF_HALFS;
    v16h af[2], bf[4];
#pragma unroll
    for (int mi = 0; mi < 2; ++mi)
      af[mi] = ldfragA(As + ((2 * wm + mi) * 16 + mrow) * AS_LD, kbase);
#pragma unroll
    for (int ni = 0; ni < 4; ++ni)
      bf[ni] = ldfragB(Bs + ((4 * wn + ni) * 16 + mrow) * AS_LD + kbB);
#pragma unroll
    for (int mi = 0; mi < 2; ++mi)
#pragma unroll
      for (int ni = 0; ni < 4; ++ni)
        acc[mi][ni] = __builtin_amdgcn_wmma_f32_16x16x32_f16(
            false, af[mi], false, bf[ni], (short)0, acc[mi][ni], false, false);
  }
#else
  // ---- fallback: register-staged pipeline (loads overlap compute) ----
  uint4 rA0 = ((const uint4*)gA)[0], rA1 = ((const uint4*)gA)[1];
  uint4 rB0 = ((const uint4*)gB)[0], rB1 = ((const uint4*)gB)[1];
  for (int kc = 0; kc < NCHUNK; ++kc) {
    const int cur = kc & 1;
    _Float16* Asw = smem + cur * 2 * BUF_HALFS;
    {
      uint4* dA = (uint4*)(Asw + ldoff);
      uint4* dB = (uint4*)(Asw + BUF_HALFS + ldoff);
      dA[0] = rA0; dA[1] = rA1;
      dB[0] = rB0; dB[1] = rB1;
    }
    __syncthreads();
    if (kc + 1 < NCHUNK) {
      const uint4* sA = (const uint4*)(gA + (kc + 1) * 32);
      const uint4* sB = (const uint4*)(gB + (kc + 1) * 32);
      rA0 = sA[0]; rA1 = sA[1];
      rB0 = sB[0]; rB1 = sB[1];
    }
    const _Float16* As = Asw;
    const _Float16* Bs = Asw + BUF_HALFS;
    v16h af[2], bf[4];
#pragma unroll
    for (int mi = 0; mi < 2; ++mi)
      af[mi] = ldfragA(As + ((2 * wm + mi) * 16 + mrow) * AS_LD, kbase);
#pragma unroll
    for (int ni = 0; ni < 4; ++ni)
      bf[ni] = ldfragB(Bs + ((4 * wn + ni) * 16 + mrow) * AS_LD + kbB);
#pragma unroll
    for (int mi = 0; mi < 2; ++mi)
#pragma unroll
      for (int ni = 0; ni < 4; ++ni)
        acc[mi][ni] = __builtin_amdgcn_wmma_f32_16x16x32_f16(
            false, af[mi], false, bf[ni], (short)0, acc[mi][ni], false, false);
  }
#endif

  __syncthreads();  // all waves done reading A/B buffers before Os aliases them

  // O' -> LDS as f16 (C/D layout: VGPR r -> M = r + 8*(lane>=16), N = lane&15)
#pragma unroll
  for (int mi = 0; mi < 2; ++mi) {
    const int mbase = (2 * wm + mi) * 16 + 8 * (lane >> 4);
#pragma unroll
    for (int ni = 0; ni < 4; ++ni) {
      const int ncol = (4 * wn + ni) * 16 + (lane & 15);
#pragma unroll
      for (int r = 0; r < 8; ++r)
        Os[(mbase + r) * OS_LD + ncol] = (_Float16)acc[mi][ni][r];
    }
  }
  __syncthreads();

  // GEMM2: out2[p=(il*4+jl)][z] = sum_{h,g} O'[il*32+h][jl*32+g] * w_out[z][h*32+g]
  const int p = lane & 15;
  const int z = wave * 16 + (lane & 15);
  const _Float16* wrow  = woutH + (size_t)z * (CH * CH) + kbB;
  const _Float16* orow0 = Os + ((p >> 2) * 32) * OS_LD + (p & 3) * 32;
  v8f c2 = {};
#pragma unroll 4
  for (int h = 0; h < 32; ++h) {
    v16h a2 = ldfragA(orow0 + h * OS_LD, kbase);
    v16h b2 = ldfragB(wrow + h * 32);
    c2 = __builtin_amdgcn_wmma_f32_16x16x32_f16(false, a2, false, b2,
                                                (short)0, c2, false, false);
  }

  const float bz = b_out[z];
#pragma unroll
  for (int r = 0; r < 8; ++r) {
    const int pp = r + 8 * (lane >> 4);
    const int i = i0 + (pp >> 2), j = j0 + (pp & 3);
    out[(((size_t)i * N_DIM) + j) * CZ + z] = c2[r] * sD[pp] + bz * sNB[pp];
  }
}

// ---------- launcher -------------------------------------------------------
extern "C" void kernel_launch(void* const* d_in, const int* in_sizes, int n_in,
                              void* d_out, int out_size, void* d_ws, size_t ws_size,
                              hipStream_t stream) {
  (void)in_sizes; (void)n_in; (void)out_size; (void)ws_size;
  const float* m     = (const float*)d_in[0];
  const float* mask  = (const float*)d_in[1];
  const float* ln_w  = (const float*)d_in[2];
  const float* ln_b  = (const float*)d_in[3];
  const float* w1    = (const float*)d_in[4];
  const float* b1    = (const float*)d_in[5];
  const float* w2    = (const float*)d_in[6];
  const float* b2    = (const float*)d_in[7];
  const float* w_out = (const float*)d_in[8];
  const float* b_out = (const float*)d_in[9];
  float* out = (float*)d_out;

  char* ws = (char*)d_ws;
  float*    normW   = (float*)(ws);                           // 256 KB
  float*    a_normW = (float*)(ws + 262144);                  // 1 KB
  _Float16* woutH   = (_Float16*)(ws + 263168);               // 256 KB
  _Float16* atT     = (_Float16*)(ws + 525312);               // 8 MB
  _Float16* btT     = (_Float16*)(ws + 525312 + 8388608);     // 8 MB

  opm_norm_kernel<<<dim3(N_DIM), dim3(256), 0, stream>>>(mask, normW, a_normW);
  opm_wout_cvt_kernel<<<dim3((CZ * CH * CH) / 256), dim3(256), 0, stream>>>(w_out, woutH);
  opm_lnproj_kernel<<<dim3((S_DIM * N_DIM) / 128), dim3(256), 0, stream>>>(
      m, mask, ln_w, ln_b, w1, b1, w2, b2, a_normW, atT, btT);
  opm_gemm_kernel<<<dim3(N_DIM / 4, N_DIM / 4), dim3(256), 0, stream>>>(
      atT, btT, woutH, normW, a_normW, b_out, out);
}
